// ParaGCNXBN2_89807766159502
// MI455X (gfx1250) — matrix operand order repaired and checked by reading, not compiled, not run on hardware
//
#include <hip/hip_runtime.h>
#include <hip/hip_bf16.h>

typedef __attribute__((ext_vector_type(16))) _Float16 v16h;
typedef __attribute__((ext_vector_type(8)))  float    v8f;

#define N_NODES 50000
#define E_RAW   800000
#define ET_     850000      // E + N, divisible by 16
#define NFEAT   128
#define NCLASS  64
#define NHEADS  8
#define HIDDEN  32

// ---------------------------------------------------------------------------
// Kernel 1: z[n, h*64+c] = sum_f x[n][f] * proj_w[h*128+f][c]
// GEMM [50000 x 128] @ [128 x 512], f16 WMMA, f32 accumulate.
// grid = (Mwaves, 32 n-tiles), block = 128 (4 waves, 1 M-tile each per iter)
// ---------------------------------------------------------------------------
__global__ __launch_bounds__(128) void zproj_kernel(
    const float* __restrict__ x, const float* __restrict__ proj_w,
    float* __restrict__ z)
{
    const int lane = threadIdx.x & 31;
    const int wv   = __builtin_amdgcn_readfirstlane(threadIdx.x >> 5); // uniform
    const int lh   = lane & 15;
    const int g    = lane >> 4;          // lane half: selects K sub-ranges
    const int nt   = blockIdx.y;         // 0..31
    const int hc   = nt * 16 + lh;       // 0..511 (head*64 + col)
    const int head = hc >> 6;
    const int col  = hc & 63;

    // B operand: 4 K-chunks of 32. ISA B layout: lanes0-15 K={0..7}U{16..23}, lanes16-31 K={8..15}U{24..31}
    v16h bz[4];
#pragma unroll
    for (int kc = 0; kc < 4; ++kc) {
#pragma unroll
        for (int i = 0; i < 16; ++i) {
            int f = kc * 32 + g * 8 + (i < 8 ? i : i + 8);
            bz[kc][i] = (_Float16)proj_w[(head * 128 + f) * NCLASS + col];
        }
    }

    const int mtiles = N_NODES / 16;     // 3125, exact
    const int stride = gridDim.x * 4;
    for (int mt = blockIdx.x * 4 + wv; mt < mtiles; mt += stride) {
        const float* xr = x + (size_t)(mt * 16 + lh) * NFEAT;
        v8f acc = {};
#pragma unroll
        for (int kc = 0; kc < 4; ++kc) {
            // ISA 16-bit A layout: g=0 -> K {0..7}U{16..23}, g=1 -> {8..15}U{24..31}
            const float* bseg = xr + kc * 32 + g * 8;
            float4 s0a = *(const float4*)(bseg);
            float4 s0b = *(const float4*)(bseg + 4);
            float4 s1a = *(const float4*)(bseg + 16);
            float4 s1b = *(const float4*)(bseg + 20);
            v16h a;
            a[0]=(_Float16)s0a.x;  a[1]=(_Float16)s0a.y;  a[2]=(_Float16)s0a.z;  a[3]=(_Float16)s0a.w;
            a[4]=(_Float16)s0b.x;  a[5]=(_Float16)s0b.y;  a[6]=(_Float16)s0b.z;  a[7]=(_Float16)s0b.w;
            a[8]=(_Float16)s1a.x;  a[9]=(_Float16)s1a.y;  a[10]=(_Float16)s1a.z; a[11]=(_Float16)s1a.w;
            a[12]=(_Float16)s1b.x; a[13]=(_Float16)s1b.y; a[14]=(_Float16)s1b.z; a[15]=(_Float16)s1b.w;
            acc = __builtin_amdgcn_wmma_f32_16x16x32_f16(
                      false, a, false, bz[kc], (short)0, acc, false, false);
        }
        // D layout: VGPR r -> row M = r + 8*g, col N = lh.
        // One base address; rows become constant instruction offsets (r*2048B).
        float* zp = z + (size_t)(mt * 16 + g * 8) * 512 + hc;
#pragma unroll
        for (int r = 0; r < 8; ++r)
            zp[r * 512] = acc[r];
    }
}

// ---------------------------------------------------------------------------
// Kernel 2: edge MLP. Per wave: 16 edges/tile, grid-stride over 53125 tiles.
//   h  = relu([x[src],x[dst]] @ w1 + b1)   (16x256 @ 256x32, f16 WMMA)
//   dyn= sigmoid(h @ w2 + b2)              (VALU via LDS bounce)
//   pe = exp(edge_weights * dyn)           (leaky_relu is identity: all > 0)
//   sums[h] += pe (global softmax denominators)
// ---------------------------------------------------------------------------
__global__ __launch_bounds__(128) void edge_mlp_kernel(
    const float* __restrict__ x, const long long* __restrict__ eidx,
    const float* __restrict__ ew,
    const float* __restrict__ w1, const float* __restrict__ b1,
    const float* __restrict__ w2, const float* __restrict__ b2,
    float* __restrict__ pe, float* __restrict__ sums)
{
    __shared__ float s_w2[HIDDEN * NHEADS];
    __shared__ float s_b2[NHEADS];
    __shared__ float s_b1[HIDDEN];
    __shared__ float s_h[4][16][HIDDEN];

    if (threadIdx.x < HIDDEN * NHEADS) s_w2[threadIdx.x] = w2[threadIdx.x];
    if (threadIdx.x < NHEADS)          s_b2[threadIdx.x] = b2[threadIdx.x];
    if (threadIdx.x < HIDDEN)          s_b1[threadIdx.x] = b1[threadIdx.x];
    __syncthreads();

    const int lane = threadIdx.x & 31;
    const int wv   = __builtin_amdgcn_readfirstlane(threadIdx.x >> 5); // uniform
    const int lh   = lane & 15;
    const int g    = lane >> 4;

    // Persistent w1 B operand: 2 N-tiles x 8 K-chunks (128 VGPRs, loaded once)
    v16h bt[2][8];
#pragma unroll
    for (int j = 0; j < 2; ++j)
#pragma unroll
        for (int kc = 0; kc < 8; ++kc)
#pragma unroll
            for (int i = 0; i < 16; ++i)
                bt[j][kc][i] =
                    (_Float16)w1[(kc * 32 + g * 8 + (i < 8 ? i : i + 8)) * HIDDEN + (j * 16 + lh)];

    float psum[NHEADS];
#pragma unroll
    for (int h = 0; h < NHEADS; ++h) psum[h] = 0.f;

    const float bias0 = s_b1[lh];
    const float bias1 = s_b1[16 + lh];

    const int ntiles  = ET_ / 16;        // 53125, exact
    const int wstride = gridDim.x * 4;
    for (int t = blockIdx.x * 4 + wv; t < ntiles; t += wstride) {
        const long long e = (long long)t * 16 + lh;   // tile never straddles E_RAW
        int srcn, dstn;
        if (e < E_RAW) { srcn = (int)eidx[e]; dstn = (int)eidx[E_RAW + e]; }
        else           { srcn = dstn = (int)(e - E_RAW); }
        const float* xs = x + (size_t)srcn * NFEAT;
        const float* xd = x + (size_t)dstn * NFEAT;

        v8f c0 = {}, c1 = {};
#pragma unroll
        for (int kc = 0; kc < 8; ++kc) {
            const float* base = (kc < 4) ? (xs + kc * 32) : (xd + (kc - 4) * 32);
            const float* bseg = base + g * 8;
            float4 s0a = *(const float4*)(bseg);
            float4 s0b = *(const float4*)(bseg + 4);
            float4 s1a = *(const float4*)(bseg + 16);
            float4 s1b = *(const float4*)(bseg + 20);
            v16h a;
            a[0]=(_Float16)s0a.x;  a[1]=(_Float16)s0a.y;  a[2]=(_Float16)s0a.z;  a[3]=(_Float16)s0a.w;
            a[4]=(_Float16)s0b.x;  a[5]=(_Float16)s0b.y;  a[6]=(_Float16)s0b.z;  a[7]=(_Float16)s0b.w;
            a[8]=(_Float16)s1a.x;  a[9]=(_Float16)s1a.y;  a[10]=(_Float16)s1a.z; a[11]=(_Float16)s1a.w;
            a[12]=(_Float16)s1b.x; a[13]=(_Float16)s1b.y; a[14]=(_Float16)s1b.z; a[15]=(_Float16)s1b.w;
            c0 = __builtin_amdgcn_wmma_f32_16x16x32_f16(
                     false, a, false, bt[0][kc], (short)0, c0, false, false);
            c1 = __builtin_amdgcn_wmma_f32_16x16x32_f16(
                     false, a, false, bt[1][kc], (short)0, c1, false, false);
        }

        // bias + relu -> LDS (D layout: row = r + 8*g, hidden col = jtile*16+lh)
#pragma unroll
        for (int r = 0; r < 8; ++r) {
            s_h[wv][g * 8 + r][lh]      = fmaxf(c0[r] + bias0, 0.f);
            s_h[wv][g * 8 + r][16 + lh] = fmaxf(c1[r] + bias1, 0.f);
        }
        __builtin_amdgcn_wave_barrier();
        asm volatile("s_wait_dscnt 0" ::: "memory");   // LDS in-order per wave

        if (lane < 16) {                 // lane l finishes edge t*16+l
            float dyn[NHEADS];
#pragma unroll
            for (int h = 0; h < NHEADS; ++h) dyn[h] = s_b2[h];
#pragma unroll
            for (int c = 0; c < HIDDEN; ++c) {
                float hv = s_h[wv][lh][c];
#pragma unroll
                for (int h = 0; h < NHEADS; ++h) dyn[h] += hv * s_w2[c * NHEADS + h];
            }
            const float* ewp = ew + e;   // head stride is constant ET_*4 bytes
            float pv[NHEADS];
#pragma unroll
            for (int h = 0; h < NHEADS; ++h) {
                float dsig = 1.f / (1.f + __expf(-dyn[h]));
                float wgt  = ewp[(size_t)h * ET_] * dsig;
                wgt  = (wgt > 0.f) ? wgt : 0.01f * wgt;   // leaky_relu (always +)
                pv[h] = __expf(wgt);
                psum[h] += pv[h];
            }
            float4* pp = (float4*)(pe + (size_t)e * 8);
            pp[0] = make_float4(pv[0], pv[1], pv[2], pv[3]);
            pp[1] = make_float4(pv[4], pv[5], pv[6], pv[7]);
        }
    }

    if (lane < 16) {
#pragma unroll
        for (int h = 0; h < NHEADS; ++h) atomicAdd(&sums[h], psum[h]);
    }
}

// ---------------------------------------------------------------------------
// Kernel 3: aggregation. One wave per edge (grid-stride):
//   acc[dst][c] += sum_h (pe[e][h]/sums[h]) * z[src][h*64 + c]
// z (102 MB) is L2-resident; 64 f32 atomics per edge.
// ---------------------------------------------------------------------------
__global__ __launch_bounds__(256) void agg_kernel(
    const long long* __restrict__ eidx, const float* __restrict__ pe,
    const float* __restrict__ sums, const float* __restrict__ z,
    float* __restrict__ acc)
{
    const int lane = threadIdx.x & 31;
    const long long wid  = (long long)blockIdx.x * 8 + (threadIdx.x >> 5);
    const long long wtot = (long long)gridDim.x * 8;
    float inv[NHEADS];
#pragma unroll
    for (int h = 0; h < NHEADS; ++h) inv[h] = 1.f / sums[h];

    for (long long e = wid; e < ET_; e += wtot) {
        int srcn, dstn;
        if (e < E_RAW) { srcn = (int)eidx[e]; dstn = (int)eidx[E_RAW + e]; }
        else           { srcn = dstn = (int)(e - E_RAW); }
        const float* pp = pe + (size_t)e * 8;
        float coef[NHEADS];
#pragma unroll
        for (int h = 0; h < NHEADS; ++h) coef[h] = pp[h] * inv[h];
        const float* zp = z + (size_t)srcn * 512 + lane;
        float v0 = 0.f, v1 = 0.f;
#pragma unroll
        for (int h = 0; h < NHEADS; ++h) {
            v0 += coef[h] * zp[h * 64];
            v1 += coef[h] * zp[h * 64 + 32];
        }
        atomicAdd(&acc[(size_t)dstn * 64 + lane], v0);
        atomicAdd(&acc[(size_t)dstn * 64 + 32 + lane], v1);
    }
}

// ---------------------------------------------------------------------------
// Kernel 4: + proj_b, LayerNorm over 64, one wave per node.
// ---------------------------------------------------------------------------
__global__ __launch_bounds__(256) void ln_kernel(
    const float* __restrict__ acc, const float* __restrict__ proj_b,
    const float* __restrict__ ln_g, const float* __restrict__ ln_b,
    float* __restrict__ out)
{
    const int lane = threadIdx.x & 31;
    const int n = blockIdx.x * 8 + (threadIdx.x >> 5);
    if (n >= N_NODES) return;
    float a0 = acc[(size_t)n * 64 + lane]      + proj_b[lane];
    float a1 = acc[(size_t)n * 64 + 32 + lane] + proj_b[32 + lane];
    float s = a0 + a1;
#pragma unroll
    for (int off = 16; off; off >>= 1) s += __shfl_xor(s, off, 32);
    float mu = s * (1.f / 64.f);
    float d0 = a0 - mu, d1 = a1 - mu;
    float q = d0 * d0 + d1 * d1;
#pragma unroll
    for (int off = 16; off; off >>= 1) q += __shfl_xor(q, off, 32);
    float rs = rsqrtf(q * (1.f / 64.f) + 1e-5f);
    out[(size_t)n * 64 + lane]      = d0 * rs * ln_g[lane]      + ln_b[lane];
    out[(size_t)n * 64 + 32 + lane] = d1 * rs * ln_g[32 + lane] + ln_b[32 + lane];
}

// ---------------------------------------------------------------------------
extern "C" void kernel_launch(void* const* d_in, const int* in_sizes, int n_in,
                              void* d_out, int out_size, void* d_ws, size_t ws_size,
                              hipStream_t stream)
{
    (void)in_sizes; (void)n_in; (void)out_size; (void)ws_size;
    const float*     x    = (const float*)d_in[0];
    const long long* eidx = (const long long*)d_in[1];   // int64 edge_index [2,E]
    const float*     ew   = (const float*)d_in[2];       // [8, ET]
    const float*     w1   = (const float*)d_in[3];
    const float*     b1   = (const float*)d_in[4];
    const float*     w2   = (const float*)d_in[5];
    const float*     b2   = (const float*)d_in[6];
    const float*     pw   = (const float*)d_in[7];
    const float*     pb   = (const float*)d_in[8];
    const float*     lg   = (const float*)d_in[9];
    const float*     lb   = (const float*)d_in[10];
    float* out = (float*)d_out;

    // workspace layout (bytes): z 102,400,000 | pe 27,200,000 | sums 256 | acc 12,800,000
    char*  ws   = (char*)d_ws;
    float* z    = (float*)(ws);
    float* pe   = (float*)(ws + 102400000ull);
    float* sums = (float*)(ws + 102400000ull + 27200000ull);
    float* acc  = (float*)(ws + 102400000ull + 27200000ull + 256ull);

    hipMemsetAsync(sums, 0, NHEADS * sizeof(float), stream);
    hipMemsetAsync(acc, 0, (size_t)N_NODES * NCLASS * sizeof(float), stream);

    zproj_kernel   <<<dim3(160, 32), 128, 0, stream>>>(x, pw, z);
    edge_mlp_kernel<<<1024,          128, 0, stream>>>(x, eidx, ew, w1, b1, w2, b2, pe, sums);
    agg_kernel     <<<2048,          256, 0, stream>>>(eidx, pe, sums, z, acc);
    ln_kernel      <<<6250,          256, 0, stream>>>(acc, pb, lg, lb, out);
}
